// Seq2seq_858993459305
// MI455X (gfx1250) — compile-verified, matrix operand order checked by reading, not run on hardware
//
#include <hip/hip_runtime.h>

#define BATCH 32
#define SEQ   512
#define EDIM  256
#define HDIM  512
#define TDEC  64
#define NL    4

typedef __attribute__((ext_vector_type(16))) _Float16 v16h;
typedef __attribute__((ext_vector_type(8)))  float    v8f;
typedef __attribute__((ext_vector_type(4)))  float    v4f;

__device__ __forceinline__ float sigf(float x) { return 1.f / (1.f + __expf(-x)); }

__device__ __forceinline__ v8f wmma16(v16h a, v16h b, v8f c) {
  // D(f32,16x16) = A(f16,16x32) * B(f16,32x16) + C
  return __builtin_amdgcn_wmma_f32_16x16x32_f16(false, a, false, b, (short)0, c, false, false);
}

// ---------------------------------------------------------------------------
// Weight pack: dst[n, 0:k0] = f16(s0[n,:]), dst[n, k0:k0+k1] = f16(s1[n,:])
// ---------------------------------------------------------------------------
__global__ void pack_w_kernel(const float* __restrict__ s0, int k0,
                              const float* __restrict__ s1, int k1,
                              _Float16* __restrict__ dst, int nrows) {
  const int K = k0 + k1;
  size_t idx = (size_t)blockIdx.x * 256 + threadIdx.x;
  size_t total = (size_t)nrows * K;
  if (idx >= total) return;
  int r = (int)(idx / K), c = (int)(idx % K);
  float v = (c < k0) ? s0[(size_t)r * k0 + c] : s1[(size_t)r * k1 + (c - k0)];
  dst[idx] = (_Float16)v;
}

// ---------------------------------------------------------------------------
// Embedding gathers (fp32 table -> f16 activations)
// ---------------------------------------------------------------------------
__global__ void embed_seq_kernel(const int* __restrict__ x, const float* __restrict__ emb,
                                 _Float16* __restrict__ xe) {
  int bs = blockIdx.x;                   // b*SEQ + s
  int b = bs / SEQ, s = bs % SEQ;
  int tok = x[(size_t)b * SEQ + s];
  xe[((size_t)s * BATCH + b) * EDIM + threadIdx.x] =
      (_Float16)emb[(size_t)tok * EDIM + threadIdx.x];
}

__global__ void embed_start_kernel(const int* __restrict__ st, const float* __restrict__ emb,
                                   _Float16* __restrict__ din /*[B,512] write cols 0:256*/) {
  int b = blockIdx.x;
  int tok = st[b];
  din[(size_t)b * 512 + threadIdx.x] = (_Float16)emb[(size_t)tok * EDIM + threadIdx.x];
}

// ---------------------------------------------------------------------------
// Persistent bidirectional LSTM layer. grid.x = 2 (dir), block = 1024 (32 waves)
// Wave w owns hidden units [16w, 16w+16): its 4 gate N-tiles + cell update.
// h kept in LDS (f16), c kept in accumulator-shaped VGPRs (f32).
// Weights stream from L2 (f16-packed, fully L2-resident); 8 waves/SIMD hide
// the L2 latency (register budget 128/wave forbids deeper SW pipelining).
// ---------------------------------------------------------------------------
__global__ __launch_bounds__(1024)
void lstm_kernel(const _Float16* __restrict__ xseq, int ins, int nsteps,
                 const _Float16* __restrict__ Wc0, const _Float16* __restrict__ Wc1,
                 const float* __restrict__ bih0, const float* __restrict__ bhh0,
                 const float* __restrict__ bih1, const float* __restrict__ bhh1,
                 float* __restrict__ hstate, float* __restrict__ cstate,
                 _Float16* __restrict__ outseq, int ostride, int init_state) {
  const int d = blockIdx.x;
  const _Float16* W = d ? Wc1 : Wc0;
  const float* bih = d ? bih1 : bih0;
  const float* bhh = d ? bhh1 : bhh0;
  float* hs = hstate + (size_t)d * BATCH * HDIM;
  float* cs = cstate + (size_t)d * BATCH * HDIM;
  const int K = ins + HDIM;
  const int tid = threadIdx.x;
  const int wave = tid >> 5, lane = tid & 31;
  const int j0 = wave * 16;
  const int r = lane & 15;               // A row within tile / B column
  const int kh = (lane >> 4) * 16;       // lane-half K offset
  const int mb = (lane < 16) ? 0 : 8;    // C/D row base per ISA f32 16x16 layout
  const int col = j0 + r;

  __shared__ _Float16 hbuf[BATCH * HDIM] __attribute__((aligned(32)));

  for (int i = tid; i < BATCH * HDIM; i += 1024)
    hbuf[i] = init_state ? (_Float16)hs[i] : (_Float16)0.f;

  v8f cA = {}, cB = {};
  if (init_state) {
    for (int v = 0; v < 8; ++v) {
      cA[v] = cs[(size_t)(mb + v) * HDIM + col];
      cB[v] = cs[(size_t)(16 + mb + v) * HDIM + col];
    }
  }
  v8f hA = {}, hB = {};

  // time-invariant gate biases (hoisted off the serial recurrence path)
  const float bi0 = bih[col] + bhh[col];
  const float bi1 = bih[HDIM + col] + bhh[HDIM + col];
  const float bi2 = bih[2 * HDIM + col] + bhh[2 * HDIM + col];
  const float bi3 = bih[3 * HDIM + col] + bhh[3 * HDIM + col];

  // per-gate weight row base pointers (per-lane)
  const _Float16* Wg0 = W + (size_t)(0 * HDIM + col) * K + kh;
  const _Float16* Wg1 = W + (size_t)(1 * HDIM + col) * K + kh;
  const _Float16* Wg2 = W + (size_t)(2 * HDIM + col) * K + kh;
  const _Float16* Wg3 = W + (size_t)(3 * HDIM + col) * K + kh;

  __syncthreads();

  for (int t = 0; t < nsteps; ++t) {
    const int ts = d ? (nsteps - 1 - t) : t;
    const _Float16* xt = xseq + (size_t)ts * BATCH * ins;

    v8f zf = {0.f, 0.f, 0.f, 0.f, 0.f, 0.f, 0.f, 0.f};
    v8f acc[4][2];
#pragma unroll
    for (int g = 0; g < 4; ++g) { acc[g][0] = zf; acc[g][1] = zf; }

    // ---- x_t @ Wih^T part (A streamed from global f16 activations) ----
    const _Float16* xa0 = xt + (size_t)r * ins + kh;
    const _Float16* xa1 = xt + (size_t)(16 + r) * ins + kh;
    for (int kk = 0; kk < ins; kk += 32) {
      v16h a0 = *(const v16h*)(xa0 + kk);
      v16h a1 = *(const v16h*)(xa1 + kk);
      __builtin_prefetch(Wg0 + kk + 128, 0, 1);
      {
        v16h b = *(const v16h*)(Wg0 + kk);
        acc[0][0] = wmma16(a0, b, acc[0][0]);
        acc[0][1] = wmma16(a1, b, acc[0][1]);
      }
      {
        v16h b = *(const v16h*)(Wg1 + kk);
        acc[1][0] = wmma16(a0, b, acc[1][0]);
        acc[1][1] = wmma16(a1, b, acc[1][1]);
      }
      {
        v16h b = *(const v16h*)(Wg2 + kk);
        acc[2][0] = wmma16(a0, b, acc[2][0]);
        acc[2][1] = wmma16(a1, b, acc[2][1]);
      }
      {
        v16h b = *(const v16h*)(Wg3 + kk);
        acc[3][0] = wmma16(a0, b, acc[3][0]);
        acc[3][1] = wmma16(a1, b, acc[3][1]);
      }
    }
    // ---- h_{t-1} @ Whh^T part (A broadcast from LDS) ----
    for (int kk = 0; kk < HDIM; kk += 32) {
      v16h a0 = *(const v16h*)(&hbuf[r * HDIM + kk + kh]);
      v16h a1 = *(const v16h*)(&hbuf[(16 + r) * HDIM + kk + kh]);
      {
        v16h b = *(const v16h*)(Wg0 + ins + kk);
        acc[0][0] = wmma16(a0, b, acc[0][0]);
        acc[0][1] = wmma16(a1, b, acc[0][1]);
      }
      {
        v16h b = *(const v16h*)(Wg1 + ins + kk);
        acc[1][0] = wmma16(a0, b, acc[1][0]);
        acc[1][1] = wmma16(a1, b, acc[1][1]);
      }
      {
        v16h b = *(const v16h*)(Wg2 + ins + kk);
        acc[2][0] = wmma16(a0, b, acc[2][0]);
        acc[2][1] = wmma16(a1, b, acc[2][1]);
      }
      {
        v16h b = *(const v16h*)(Wg3 + ins + kk);
        acc[3][0] = wmma16(a0, b, acc[3][0]);
        acc[3][1] = wmma16(a1, b, acc[3][1]);
      }
    }
    __syncthreads();  // all h reads done before h update

#pragma unroll
    for (int mt = 0; mt < 2; ++mt) {
#pragma unroll
      for (int v = 0; v < 8; ++v) {
        int m = mt * 16 + mb + v;
        float iv = sigf(acc[0][mt][v] + bi0);
        float fv = sigf(acc[1][mt][v] + bi1);
        float gv = tanhf(acc[2][mt][v] + bi2);
        float ov = sigf(acc[3][mt][v] + bi3);
        float cp = mt ? cB[v] : cA[v];
        float cn = fv * cp + iv * gv;
        float hn = ov * tanhf(cn);
        if (mt) { cB[v] = cn; hB[v] = hn; } else { cA[v] = cn; hA[v] = hn; }
        hbuf[m * HDIM + col] = (_Float16)hn;
        outseq[(size_t)ts * BATCH * ostride + (size_t)m * ostride + d * HDIM + col] =
            (_Float16)hn;
      }
    }
    __syncthreads();  // h update visible before next step
  }

  // final (h, c) fp32 states -> feed decoder / carry across decoder steps
  for (int v = 0; v < 8; ++v) {
    hs[(size_t)(mb + v) * HDIM + col]      = hA[v];
    hs[(size_t)(16 + mb + v) * HDIM + col] = hB[v];
    cs[(size_t)(mb + v) * HDIM + col]      = cA[v];
    cs[(size_t)(16 + mb + v) * HDIM + col] = cB[v];
  }
}

// ---------------------------------------------------------------------------
// Generic WMMA GEMM: C[M,N] = act(A[M,K]_f16 @ W[N,K]^T_f16 + bias)
// grid.x = N/16, grid.y = M/32, block = 32 (one wave, 32x16 output tile)
// ---------------------------------------------------------------------------
__global__ __launch_bounds__(32)
void gemm_kernel(const _Float16* __restrict__ A, int lda,
                 const _Float16* __restrict__ W,
                 const float* __restrict__ bias,
                 float* __restrict__ C32, int ldc32,
                 _Float16* __restrict__ C16, int ldc16,
                 int K, int act) {
  const int lane = threadIdx.x;
  const int n0 = blockIdx.x * 16;
  const int m0 = blockIdx.y * 32;
  const int r = lane & 15;
  const int kh = (lane >> 4) * 16;
  const _Float16* pa0 = A + (size_t)(m0 + r) * lda + kh;
  const _Float16* pa1 = A + (size_t)(m0 + 16 + r) * lda + kh;
  const _Float16* pb  = W + (size_t)(n0 + r) * K + kh;
  v8f acc0 = {}, acc1 = {};
  for (int kk = 0; kk < K; kk += 32) {
    v16h a0 = *(const v16h*)(pa0 + kk);
    v16h a1 = *(const v16h*)(pa1 + kk);
    v16h b  = *(const v16h*)(pb + kk);
    acc0 = wmma16(a0, b, acc0);
    acc1 = wmma16(a1, b, acc1);
  }
  const int n = n0 + r;
  const int mb = (lane < 16) ? 0 : 8;
  const float bv = bias ? bias[n] : 0.f;
#pragma unroll
  for (int v = 0; v < 8; ++v) {
    float x0 = acc0[v] + bv, x1 = acc1[v] + bv;
    if (act) { x0 = tanhf(x0); x1 = tanhf(x1); }
    int ra = m0 + mb + v, rb = m0 + 16 + mb + v;
    if (C32) { C32[(size_t)ra * ldc32 + n] = x0; C32[(size_t)rb * ldc32 + n] = x1; }
    if (C16) { C16[(size_t)ra * ldc16 + n] = (_Float16)x0;
               C16[(size_t)rb * ldc16 + n] = (_Float16)x1; }
  }
}

// ---------------------------------------------------------------------------
// Attention core: scores (batched GEMV) + softmax + mix. One block per batch.
// qv fp32 [B,E]; ctx fp32 [S,B,E]; mix f16 -> mix_out[b*ldm + e]
// q staged in LDS; score dot vectorized with float4 (b128 loads).
// ---------------------------------------------------------------------------
__global__ __launch_bounds__(256)
void attn_kernel(const float* __restrict__ qv, const float* __restrict__ ctx,
                 _Float16* __restrict__ mix_out, int ldm) {
  const int b = blockIdx.x, tid = threadIdx.x;
  __shared__ float sc[SEQ];
  __shared__ float red[256];
  __shared__ float qsh[EDIM] __attribute__((aligned(16)));
  qsh[tid] = qv[(size_t)b * EDIM + tid];
  __syncthreads();
  for (int s = tid; s < SEQ; s += 256) {
    const v4f* cp = (const v4f*)(ctx + ((size_t)s * BATCH + b) * EDIM);
    const v4f* qp = (const v4f*)qsh;
    float acc = 0.f;
#pragma unroll 4
    for (int e = 0; e < EDIM / 4; ++e) {
      v4f c4 = cp[e], q4 = qp[e];
      acc += q4.x * c4.x + q4.y * c4.y + q4.z * c4.z + q4.w * c4.w;
    }
    sc[s] = acc;
  }
  __syncthreads();
  red[tid] = fmaxf(sc[tid], sc[tid + 256]);
  __syncthreads();
  for (int o = 128; o > 0; o >>= 1) { if (tid < o) red[tid] = fmaxf(red[tid], red[tid + o]); __syncthreads(); }
  float mx = red[0];
  __syncthreads();
  float e0 = __expf(sc[tid] - mx), e1 = __expf(sc[tid + 256] - mx);
  sc[tid] = e0; sc[tid + 256] = e1;
  red[tid] = e0 + e1;
  __syncthreads();
  for (int o = 128; o > 0; o >>= 1) { if (tid < o) red[tid] += red[tid + o]; __syncthreads(); }
  float inv = 1.f / red[0];
  __syncthreads();
  float acc = 0.f;
  for (int s = 0; s < SEQ; ++s) acc += sc[s] * ctx[((size_t)s * BATCH + b) * EDIM + tid];
  mix_out[(size_t)b * ldm + tid] = (_Float16)(acc * inv);
}

// grade[b,t,:] = dout[b] @ grade_W^T + grade_b  (N=2: VALU)
__global__ void grade_kernel(const float* __restrict__ dout, const float* __restrict__ gW,
                             const float* __restrict__ gb, float* __restrict__ out, int t) {
  int tid = threadIdx.x;
  if (tid >= 64) return;
  int b = tid >> 1, g = tid & 1;
  float acc = gb[g];
  for (int e = 0; e < EDIM; ++e) acc += dout[(size_t)b * EDIM + e] * gW[(size_t)g * EDIM + e];
  out[((size_t)b * TDEC + t) * 2 + g] = acc;
}

// ---------------------------------------------------------------------------
extern "C" void kernel_launch(void* const* d_in, const int* in_sizes, int n_in,
                              void* d_out, int out_size, void* d_ws, size_t ws_size,
                              hipStream_t stream) {
  (void)in_sizes; (void)n_in; (void)out_size; (void)ws_size;
  const int* x            = (const int*)d_in[0];
  const int* start_tokens = (const int*)d_in[3];
  const float* embedding  = (const float*)d_in[4];
  // enc_params: 5..36, dec_params: 37..68  ((l*2+d)*4 + {Wih,Whh,bih,bhh})
  const float* encP[NL][2][4];
  const float* decP[NL][2][4];
  for (int l = 0; l < NL; ++l)
    for (int d = 0; d < 2; ++d)
      for (int k = 0; k < 4; ++k) {
        encP[l][d][k] = (const float*)d_in[5  + (l * 2 + d) * 4 + k];
        decP[l][d][k] = (const float*)d_in[37 + (l * 2 + d) * 4 + k];
      }
  const float* enc_W   = (const float*)d_in[69];
  const float* enc_b   = (const float*)d_in[70];
  const float* dec_W   = (const float*)d_in[71];
  const float* dec_b   = (const float*)d_in[72];
  const float* attn_Win  = (const float*)d_in[73];
  const float* attn_Wout = (const float*)d_in[74];
  const float* grade_W = (const float*)d_in[75];
  const float* grade_b = (const float*)d_in[76];
  float* out = (float*)d_out;

  // ---- workspace carve-out (256B aligned) ----
  size_t off = 0;
  char* base = (char*)d_ws;
  auto carve = [&](size_t bytes) -> void* {
    off = (off + 255) & ~(size_t)255;
    void* p = base + off;
    off += bytes;
    return p;
  };
  _Float16* xe   = (_Float16*)carve((size_t)SEQ * BATCH * EDIM * 2);       // [S,B,E]
  _Float16* seqA = (_Float16*)carve((size_t)SEQ * BATCH * 1024 * 2);       // ping
  _Float16* seqB = (_Float16*)carve((size_t)SEQ * BATCH * 1024 * 2);       // pong
  _Float16* encWc[NL][2];
  _Float16* decWc[NL][2];
  for (int l = 0; l < NL; ++l)
    for (int d = 0; d < 2; ++d) {
      int ki = (l == 0) ? EDIM : 1024;
      encWc[l][d] = (_Float16*)carve((size_t)2048 * (ki + HDIM) * 2);
      int kd = (l == 0) ? 512 : 1024;
      decWc[l][d] = (_Float16*)carve((size_t)2048 * (kd + HDIM) * 2);
    }
  _Float16* encW16  = (_Float16*)carve((size_t)EDIM * 1024 * 2);
  _Float16* decW16  = (_Float16*)carve((size_t)EDIM * 1024 * 2);
  _Float16* Win16   = (_Float16*)carve((size_t)EDIM * EDIM * 2);
  _Float16* Wout16  = (_Float16*)carve((size_t)EDIM * 512 * 2);
  float*    context = (float*)   carve((size_t)SEQ * BATCH * EDIM * 4);    // [S,B,E]
  float*    hstate  = (float*)   carve((size_t)2 * NL * BATCH * HDIM * 4);
  float*    cstate  = (float*)   carve((size_t)2 * NL * BATCH * HDIM * 4);
  float*    dout32  = (float*)   carve((size_t)BATCH * EDIM * 4);
  float*    q32     = (float*)   carve((size_t)BATCH * EDIM * 4);
  _Float16* din     = (_Float16*)carve((size_t)BATCH * 512 * 2);   // [dout|a]
  _Float16* catqm   = (_Float16*)carve((size_t)BATCH * 512 * 2);   // [mix|q]
  _Float16* dpp0    = (_Float16*)carve((size_t)BATCH * 1024 * 2);
  _Float16* dpp1    = (_Float16*)carve((size_t)BATCH * 1024 * 2);

  auto pack = [&](const float* s0, int k0, const float* s1, int k1, _Float16* dst, int nrows) {
    size_t total = (size_t)nrows * (k0 + k1);
    int blocks = (int)((total + 255) / 256);
    pack_w_kernel<<<blocks, 256, 0, stream>>>(s0, k0, s1, k1, dst, nrows);
  };

  // ---- pack all weights to f16 (L2-resident) ----
  for (int l = 0; l < NL; ++l)
    for (int d = 0; d < 2; ++d) {
      int ki = (l == 0) ? EDIM : 1024;
      pack(encP[l][d][0], ki, encP[l][d][1], HDIM, encWc[l][d], 2048);
      int kd = (l == 0) ? 512 : 1024;
      pack(decP[l][d][0], kd, decP[l][d][1], HDIM, decWc[l][d], 2048);
    }
  pack(enc_W, 1024, nullptr, 0, encW16, EDIM);
  pack(dec_W, 1024, nullptr, 0, decW16, EDIM);
  pack(attn_Win, EDIM, nullptr, 0, Win16, EDIM);
  pack(attn_Wout, 512, nullptr, 0, Wout16, EDIM);

  // ---- encoder ----
  embed_seq_kernel<<<BATCH * SEQ, EDIM, 0, stream>>>(x, embedding, xe);
  {
    const _Float16* in = xe;  int ins = EDIM;  _Float16* outp = seqA;
    for (int l = 0; l < NL; ++l) {
      lstm_kernel<<<2, 1024, 0, stream>>>(
          in, ins, SEQ, encWc[l][0], encWc[l][1],
          encP[l][0][2], encP[l][0][3], encP[l][1][2], encP[l][1][3],
          hstate + (size_t)(2 * l) * BATCH * HDIM,
          cstate + (size_t)(2 * l) * BATCH * HDIM,
          outp, 1024, /*init_state=*/0);
      in = outp; ins = 1024;
      outp = (outp == seqA) ? seqB : seqA;
    }
    // context = enc_rnn @ enc_W^T + enc_b   [16384,1024]x[1024,256]
    gemm_kernel<<<dim3(EDIM / 16, (SEQ * BATCH) / 32), 32, 0, stream>>>(
        in, 1024, encW16, enc_b, context, EDIM, nullptr, 0, 1024, /*act=*/0);
  }

  // ---- decoder init: din = [embed(start) | attn(embed(start))] ----
  embed_start_kernel<<<BATCH, EDIM, 0, stream>>>(start_tokens, embedding, din);
  gemm_kernel<<<dim3(EDIM / 16, 1), 32, 0, stream>>>(
      din, 512, Win16, nullptr, q32, EDIM, catqm + EDIM, 512, EDIM, 0);
  attn_kernel<<<BATCH, 256, 0, stream>>>(q32, context, catqm, 512);
  gemm_kernel<<<dim3(EDIM / 16, 1), 32, 0, stream>>>(
      catqm, 512, Wout16, nullptr, nullptr, 0, din + EDIM, 512, 512, /*tanh=*/1);

  // ---- decoder loop ----
  for (int t = 0; t < TDEC; ++t) {
    const _Float16* lin = din; int ins = 512;
    _Float16* lout = dpp0;
    for (int l = 0; l < NL; ++l) {
      lstm_kernel<<<2, 1024, 0, stream>>>(
          lin, ins, 1, decWc[l][0], decWc[l][1],
          decP[l][0][2], decP[l][0][3], decP[l][1][2], decP[l][1][3],
          hstate + (size_t)(2 * l) * BATCH * HDIM,
          cstate + (size_t)(2 * l) * BATCH * HDIM,
          lout, 1024, /*init_state=*/1);
      lin = lout; ins = 1024;
      lout = (lout == dpp0) ? dpp1 : dpp0;
    }
    // dout = rnn_out @ dec_W^T + dec_b ; f16 copy into din[:,0:256]
    gemm_kernel<<<dim3(EDIM / 16, 1), 32, 0, stream>>>(
        lin, 1024, decW16, dec_b, dout32, EDIM, din, 512, 1024, 0);
    grade_kernel<<<1, 64, 0, stream>>>(dout32, grade_W, grade_b, out, t);
    // q = dout @ Win^T ; f16 copy into catqm[:,256:512]
    gemm_kernel<<<dim3(EDIM / 16, 1), 32, 0, stream>>>(
        din, 512, Win16, nullptr, q32, EDIM, catqm + EDIM, 512, EDIM, 0);
    attn_kernel<<<BATCH, 256, 0, stream>>>(q32, context, catqm, 512);
    // a = tanh([mix|q] @ Wout^T) -> din[:,256:512]
    gemm_kernel<<<dim3(EDIM / 16, 1), 32, 0, stream>>>(
        catqm, 512, Wout16, nullptr, nullptr, 0, din + EDIM, 512, 512, 1);
  }
}
